// OutputBlock_30932354466267
// MI455X (gfx1250) — matrix operand order repaired and checked by reading, not compiled, not run on hardware
//
#include <hip/hip_runtime.h>
#include <math.h>

// ---------------------------------------------------------------------------
// Problem constants (match reference)
// ---------------------------------------------------------------------------
#define HIDDEN      128
#define OUTDIM      256
#define NUM_RADIAL  6
#define NUM_LAYERS  3

// MLP tiling
#define M_TILE      32          // node rows per workgroup
#define LDB         264         // padded LDS row stride (bf16 elems); 264*2B=528B, 132 dw, %64=4
#define THREADS     256         // 8 waves (wave32)

typedef __attribute__((ext_vector_type(2)))  float  v2f;
typedef __attribute__((ext_vector_type(4)))  float  v4f;
typedef __attribute__((ext_vector_type(8)))  float  v8f;
typedef __attribute__((ext_vector_type(16))) __bf16 v16bf;

union frag16 {            // pun two 16B loads into one WMMA operand
    v16bf v;
    v4f   f[2];
};

__device__ __forceinline__ void split_bf16(float v, __bf16& hi, __bf16& lo) {
    hi = (__bf16)v;                 // RNE
    lo = (__bf16)(v - (float)hi);   // v - (float)hi is exact in f32
}

// ---------------------------------------------------------------------------
// Kernel 1: zero the node accumulator (must re-zero every replay)
// ---------------------------------------------------------------------------
__global__ void zero_kernel(float* __restrict__ p, long long n4) {
    long long i = (long long)blockIdx.x * blockDim.x + threadIdx.x;
    long long stride = (long long)gridDim.x * blockDim.x;
    v4f z = {0.f, 0.f, 0.f, 0.f};
    for (; i < n4; i += stride)
        reinterpret_cast<v4f*>(p)[i] = z;
}

// ---------------------------------------------------------------------------
// Kernel 2: split f32 weights into bf16 hi/lo planes in a contiguous-k
// packed-B layout.  For a K x 256 matrix, packed index
//   p = ((kt*256 + col)*2 + half)*16 + j   holds  W[kt*32 + half*16 + j][col]
// so one lane's B fragment is 16 contiguous bf16 (32 B) -> 2x b128 loads.
// ---------------------------------------------------------------------------
__global__ void split_pack_kernel(const float* __restrict__ W, int K,
                                  __bf16* __restrict__ hi, __bf16* __restrict__ lo) {
    const int total = K * OUTDIM;
    int p = blockIdx.x * blockDim.x + threadIdx.x;
    const int stride = gridDim.x * blockDim.x;
    for (; p < total; p += stride) {
        const int j    = p & 15;
        const int half = (p >> 4) & 1;
        const int col  = (p >> 5) & 255;
        const int kt   = p >> 13;
        const int k    = kt * 32 + half * 16 + j;
        const float v  = W[(size_t)k * OUTDIM + col];
        __bf16 h, l;
        split_bf16(v, h, l);
        hi[p] = h;
        lo[p] = l;
    }
}

// ---------------------------------------------------------------------------
// Kernel 3: g = (rbf @ W_rbf) * x, scatter-add into hsum[node][128].
// One wave per edge; lane handles 4 channels. 25.6 MB accumulator is
// L2-resident (192 MB), so global_atomic_add_f32 resolves in L2.
// ---------------------------------------------------------------------------
__global__ void edge_scatter_kernel(const float* __restrict__ x,
                                    const float* __restrict__ rbf,
                                    const int*   __restrict__ idx,
                                    const float* __restrict__ W_rbf,
                                    float*       __restrict__ hsum,
                                    int E) {
    __shared__ float Wlds[NUM_RADIAL * HIDDEN];   // 3 KB
    for (int k = threadIdx.x; k < NUM_RADIAL * HIDDEN; k += blockDim.x)
        Wlds[k] = W_rbf[k];
    __syncthreads();

    const int lane  = threadIdx.x & 31;
    const int gwave = (blockIdx.x * blockDim.x + threadIdx.x) >> 5;
    const int nwave = (gridDim.x * blockDim.x) >> 5;
    const int c = lane * 4;

    for (int e = gwave; e < E; e += nwave) {
        float r[NUM_RADIAL];
#pragma unroll
        for (int j = 0; j < NUM_RADIAL; ++j)
            r[j] = rbf[(size_t)e * NUM_RADIAL + j];   // broadcast loads, L0-hot

        const v4f xv = *reinterpret_cast<const v4f*>(x + (size_t)e * HIDDEN + c);

        float g[4];
#pragma unroll
        for (int t = 0; t < 4; ++t) {
            float w = 0.f;
#pragma unroll
            for (int j = 0; j < NUM_RADIAL; ++j)
                w += r[j] * Wlds[j * HIDDEN + c + t];
            g[t] = w * xv[t];
        }

        float* hp = hsum + (size_t)idx[e] * HIDDEN + c;
        atomicAdd(hp + 0, g[0]);
        atomicAdd(hp + 1, g[1]);
        atomicAdd(hp + 2, g[2]);
        atomicAdd(hp + 3, g[3]);
    }
}

// ---------------------------------------------------------------------------
// One GEMM layer, bf16x3 split:  acc += Ahi*Bhi + Ahi*Blo + Alo*Bhi
// using V_WMMA_F32_16X16X32_BF16 (f32 accumulate).
//   A (16x32 bf16): lane<16 -> K = k0+0..15; lane>=16 -> k0+16..31
//   B packed as produced by split_pack_kernel (16 contiguous bf16 / lane)
//   C/D 16x16 f32: VGPR j -> row j / j+8, col = lane&15
// If gout != nullptr the epilogue writes f32 straight to global (last layer),
// otherwise it re-splits to the LDS hi/lo destination planes.
// ---------------------------------------------------------------------------
__device__ __forceinline__ void gemm_layer_bf16(
        const __bf16* __restrict__ srcHi, const __bf16* __restrict__ srcLo, // LDS, stride LDB
        const __bf16* __restrict__ WHi,   const __bf16* __restrict__ WLo,   // packed global
        const float*  __restrict__ bias,                                    // nullptr if none
        int Kt, bool doSilu,
        __bf16* __restrict__ dstHi, __bf16* __restrict__ dstLo,             // LDS (if gout==0)
        float* __restrict__ gout, int node0, int N,                         // global (if !=0)
        int wm, int wn, int lane) {
    const int half = lane >> 4;
    const int l16  = lane & 15;
    const int rowA = wm * 16 + l16;

    v8f acc[4] = {v8f{}, v8f{}, v8f{}, v8f{}};

    for (int kt = 0; kt < Kt; ++kt) {
        const int k0 = kt * 32;
        const int aoff = rowA * LDB + k0 + half * 16;

        frag16 Ahi, Alo;
        Ahi.f[0] = *reinterpret_cast<const v4f*>(srcHi + aoff);
        Ahi.f[1] = *reinterpret_cast<const v4f*>(srcHi + aoff + 8);
        Alo.f[0] = *reinterpret_cast<const v4f*>(srcLo + aoff);
        Alo.f[1] = *reinterpret_cast<const v4f*>(srcLo + aoff + 8);

#pragma unroll
        for (int t = 0; t < 4; ++t) {
            const int col  = wn * 64 + t * 16 + l16;
            const size_t p = ((size_t)(kt * 256 + col) * 2 + half) * 16;

            frag16 Bhi, Blo;
            Bhi.f[0] = *reinterpret_cast<const v4f*>(WHi + p);
            Bhi.f[1] = *reinterpret_cast<const v4f*>(WHi + p + 8);
            Blo.f[0] = *reinterpret_cast<const v4f*>(WLo + p);
            Blo.f[1] = *reinterpret_cast<const v4f*>(WLo + p + 8);

            acc[t] = __builtin_amdgcn_wmma_f32_16x16x32_bf16(
                false, Ahi.v, false, Bhi.v, (short)0, acc[t], false, false);
            acc[t] = __builtin_amdgcn_wmma_f32_16x16x32_bf16(
                false, Ahi.v, false, Blo.v, (short)0, acc[t], false, false);
            acc[t] = __builtin_amdgcn_wmma_f32_16x16x32_bf16(
                false, Alo.v, false, Bhi.v, (short)0, acc[t], false, false);
        }
    }

#pragma unroll
    for (int t = 0; t < 4; ++t) {
        const int col = wn * 64 + t * 16 + l16;
        const float bv = bias ? bias[col] : 0.f;
#pragma unroll
        for (int j = 0; j < 8; ++j) {
            float v = acc[t][j] + bv;
            if (doSilu)
                v = v * (1.0f / (1.0f + __expf(-v)));   // SiLU
            const int row = wm * 16 + half * 8 + j;
            if (gout) {
                const int node = node0 + row;
                if (node < N)
                    gout[(size_t)node * OUTDIM + col] = v;   // f32, no bf16 loss
            } else {
                __bf16 h, l;
                split_bf16(v, h, l);
                dstHi[row * LDB + col] = h;
                dstLo[row * LDB + col] = l;
            }
        }
    }
}

// ---------------------------------------------------------------------------
// Kernel 4: fused  h@W_down  ->  3x SiLU(h@Ws[l]+bs[l]).
// 32-row activation tile lives in LDS as bf16 hi/lo planes (ping-pong),
// never spilled to HBM.  Weights stream from L2 (0.9 MB hot).
// ---------------------------------------------------------------------------
__global__ __launch_bounds__(THREADS)
void node_mlp_kernel(const float*  __restrict__ hsum,    // [N][128] f32
                     const __bf16* __restrict__ WdHi, const __bf16* __restrict__ WdLo,
                     const __bf16* __restrict__ WsHi, const __bf16* __restrict__ WsLo,
                     const float*  __restrict__ bs,      // [3][256]
                     float*        __restrict__ out,     // [N][256] f32
                     int N) {
    __shared__ __bf16 aHi[2][M_TILE * LDB];   // 16.9 KB each
    __shared__ __bf16 aLo[2][M_TILE * LDB];

    const int tid    = threadIdx.x;
    const int lane   = tid & 31;
    const int waveId = tid >> 5;
    const int wm     = waveId >> 2;   // 0..1  (row tile of 16)
    const int wn     = waveId & 3;    // 0..3  (col tile of 64)
    const int node0  = blockIdx.x * M_TILE;

    // ---- load h tile [32][128], split to bf16 hi/lo planes (buffer 0) ----
#pragma unroll
    for (int q = 0; q < 4; ++q) {
        const int li   = tid + q * THREADS;   // float4 index, 32*32 of them
        const int row  = li >> 5;             // 32 float4 per row
        const int c4   = (li & 31) * 4;
        const int node = node0 + row;
        v4f v = {0.f, 0.f, 0.f, 0.f};
        if (node < N)
            v = *reinterpret_cast<const v4f*>(hsum + (size_t)node * HIDDEN + c4);
#pragma unroll
        for (int s = 0; s < 4; ++s) {
            __bf16 h, l;
            split_bf16(v[s], h, l);
            aHi[0][row * LDB + c4 + s] = h;
            aLo[0][row * LDB + c4 + s] = l;
        }
    }
    __syncthreads();

    // ---- layer 0: [32,128] @ W_down -> buffer 1 (no bias / act) ----
    gemm_layer_bf16(aHi[0], aLo[0], WdHi, WdLo, nullptr, HIDDEN / 32, false,
                    aHi[1], aLo[1], nullptr, node0, N, wm, wn, lane);
    __syncthreads();

    // ---- layer 1: SiLU(h @ Ws[0] + bs[0]) -> buffer 0 ----
    gemm_layer_bf16(aHi[1], aLo[1], WsHi, WsLo, bs, OUTDIM / 32, true,
                    aHi[0], aLo[0], nullptr, node0, N, wm, wn, lane);
    __syncthreads();

    // ---- layer 2: SiLU(h @ Ws[1] + bs[1]) -> buffer 1 ----
    gemm_layer_bf16(aHi[0], aLo[0], WsHi + (size_t)OUTDIM * OUTDIM,
                    WsLo + (size_t)OUTDIM * OUTDIM, bs + OUTDIM, OUTDIM / 32, true,
                    aHi[1], aLo[1], nullptr, node0, N, wm, wn, lane);
    __syncthreads();

    // ---- layer 3: SiLU(h @ Ws[2] + bs[2]) -> global out (f32) ----
    gemm_layer_bf16(aHi[1], aLo[1], WsHi + (size_t)2 * OUTDIM * OUTDIM,
                    WsLo + (size_t)2 * OUTDIM * OUTDIM, bs + 2 * OUTDIM, OUTDIM / 32, true,
                    nullptr, nullptr, out, node0, N, wm, wn, lane);
}

// ---------------------------------------------------------------------------
// Launch
// Inputs: 0:x[E,128] 1:rbf[E,6] 2:i[E] 3:W_rbf[6,128] 4:W_down[128,256]
//         5:Ws[3,256,256] 6:bs[3,256] 7:num_nodes(scalar)
// Workspace: [ hsum f32 N*128 | WdHi | WdLo | WsHi | WsLo ]
// ---------------------------------------------------------------------------
extern "C" void kernel_launch(void* const* d_in, const int* in_sizes, int n_in,
                              void* d_out, int out_size, void* d_ws, size_t ws_size,
                              hipStream_t stream) {
    const float* x      = (const float*)d_in[0];
    const float* rbf    = (const float*)d_in[1];
    const int*   idx    = (const int*)d_in[2];
    const float* W_rbf  = (const float*)d_in[3];
    const float* W_down = (const float*)d_in[4];
    const float* Ws     = (const float*)d_in[5];
    const float* bs     = (const float*)d_in[6];

    const int E = in_sizes[2];
    const int N = out_size / OUTDIM;

    char* ws = (char*)d_ws;
    float*  hsum = (float*)ws;                         ws += (size_t)N * HIDDEN * sizeof(float);
    __bf16* WdHi = (__bf16*)ws;                        ws += (size_t)HIDDEN * OUTDIM * sizeof(__bf16);
    __bf16* WdLo = (__bf16*)ws;                        ws += (size_t)HIDDEN * OUTDIM * sizeof(__bf16);
    __bf16* WsHi = (__bf16*)ws;                        ws += (size_t)NUM_LAYERS * OUTDIM * OUTDIM * sizeof(__bf16);
    __bf16* WsLo = (__bf16*)ws;
    float* out = (float*)d_out;

    // 1) zero accumulator
    {
        long long n4 = (long long)N * HIDDEN / 4;
        int blocks = (int)((n4 + THREADS - 1) / THREADS);
        zero_kernel<<<blocks, THREADS, 0, stream>>>(hsum, n4);
    }
    // 2) split + pack weights to bf16 hi/lo (W_down: K=128; Ws as one K=768 stack)
    split_pack_kernel<<<128, THREADS, 0, stream>>>(W_down, HIDDEN, WdHi, WdLo);
    split_pack_kernel<<<768, THREADS, 0, stream>>>(Ws, NUM_LAYERS * OUTDIM, WsHi, WsLo);
    // 3) edge gate + scatter-add
    edge_scatter_kernel<<<2048, THREADS, 0, stream>>>(x, rbf, idx, W_rbf, hsum, E);
    // 4) fused node MLP (bf16x3 split WMMA, f32 accumulate)
    {
        int blocks = (N + M_TILE - 1) / M_TILE;
        node_mlp_kernel<<<blocks, THREADS, 0, stream>>>(hsum, WdHi, WdLo, WsHi, WsLo,
                                                        bs, out, N);
    }
}
